// Model_34127810134526
// MI455X (gfx1250) — matrix operand order repaired and checked by reading, not compiled, not run on hardware
//
#include <hip/hip_runtime.h>
#include <hip/hip_bf16.h>
#include <hip/hip_fp16.h>

// ---------------------------------------------------------------------------
// MI455X / gfx1250 implementation. All heavy math through
// v_wmma_f32_16x16x32_f16 (f16 inputs, f32 accumulate).
// ---------------------------------------------------------------------------

typedef _Float16 v8h  __attribute__((ext_vector_type(8)));
typedef _Float16 v16h __attribute__((ext_vector_type(16)));
typedef float    v8f  __attribute__((ext_vector_type(8)));

#define NSTK   1024
#define SEQT   32
#define DFEAT  158
#define KPAD   160            // 158 padded to a multiple of 32
#define DM     256
#define NROWS  (NSTK*SEQT)    // 32768

// A fragment 16x32 f16: lane holds row (lane&15); half-wave
// selects 16 contiguous K starting at hi*16 (matches B).
__device__ __forceinline__ v16h ldfragA(const _Float16* rowp, int hi) {
    v16h a;
    const _Float16* p = rowp + hi * 16;
    ((uint4*)&a)[0] = *(const uint4*)(p);
    ((uint4*)&a)[1] = *(const uint4*)(p + 8);
    return a;
}
// B fragment 32x16 f16: lane holds column (lane&15); 16 contiguous K halfs
// starting at hi*16 (caller bakes hi*16 into the pointer).
__device__ __forceinline__ v16h ldfragB(const _Float16* p) {
    v16h b;
    ((uint4*)&b)[0] = *(const uint4*)(p);
    ((uint4*)&b)[1] = *(const uint4*)(p + 8);
    return b;
}

// Order-preserving float <-> uint encoding for integer atomic max.
__device__ __forceinline__ unsigned enc_f32(float x) {
    int b = __float_as_int(x);
    return (unsigned)(b ^ ((b >> 31) | 0x80000000));
}
__device__ __forceinline__ float dec_f32(unsigned k) {
    int kk = (int)k;
    return __int_as_float((kk < 0) ? (kk ^ 0x80000000) : ~kk);
}

// ===========================================================================
// Conversion kernels (f32 -> f16, weights transposed to [N][K])
// ===========================================================================
__global__ __launch_bounds__(256) void k_pad_x(const float* __restrict__ x,
                                               _Float16* __restrict__ xp) {
    int i = blockIdx.x * 256 + threadIdx.x;          // over NROWS*KPAD
    if (i >= NROWS * KPAD) return;
    int r = i / KPAD, k = i - r * KPAD;
    xp[i] = (_Float16)((k < DFEAT) ? x[r * DFEAT + k] : 0.0f);
}

__global__ __launch_bounds__(256) void k_wt_pad(const float* __restrict__ w,
                                                _Float16* __restrict__ wt) {
    // emb_w [158,256] -> wt [256,160] (transposed, zero padded K)
    int i = blockIdx.x * 256 + threadIdx.x;          // 256*160
    if (i >= DM * KPAD) return;
    int n = i / KPAD, k = i - n * KPAD;
    wt[i] = (_Float16)((k < DFEAT) ? w[k * DM + n] : 0.0f);
}

__global__ __launch_bounds__(256) void k_wt(const float* __restrict__ w,
                                            _Float16* __restrict__ wt) {
    // [256,256] -> transposed f16 [256,256]
    int i = blockIdx.x * 256 + threadIdx.x;
    if (i >= DM * DM) return;
    int n = i >> 8, k = i & 255;
    wt[i] = (_Float16)w[k * DM + n];
}

// Sequence-major transpose of V for the attention P@V WMMA:
// Vt[(b*256 + col)*seqLen + s] = V[(b*bStride + s*sStride)*256 + col]
__global__ __launch_bounds__(256) void k_vt(const _Float16* __restrict__ V,
                                            _Float16* __restrict__ Vt,
                                            int seqLen, int sStride, int bStride,
                                            int sdShift) {
    int i = blockIdx.x * 256 + threadIdx.x;          // over NROWS*DM
    int b   = i >> sdShift;
    int s   = (i >> 8) & (seqLen - 1);
    int col = i & 255;
    Vt[((size_t)((b << 8) + col)) * seqLen + s] =
        V[(size_t)(b * bStride + s * sStride) * DM + col];
}

// ===========================================================================
// WMMA GEMM: out[M,256] = A[M,KA](f16) @ Wt[N,KA]^T (+bias)(+PE)(ReLU)(+res)
// 8 waves/block; wave computes 32 rows x 64 cols (8 accum tiles).
// KA is a template constant: fully unrolled, immediate-offset addressing.
// ===========================================================================
template<int KA, bool BIAS, bool PE, bool RELU, bool RES>
__global__ __launch_bounds__(256) void k_gemm(const _Float16* __restrict__ A,
                                              const _Float16* __restrict__ Wt,
                                              const float* __restrict__ bias,
                                              const float* __restrict__ pe,
                                              const _Float16* __restrict__ res,
                                              _Float16* __restrict__ out) {
    const int lane = threadIdx.x & 31;
    const int wv   = threadIdx.x >> 5;
    const int rb   = blockIdx.x * 256 + wv * 32;     // row base (M), 32 rows
    const int cb   = blockIdx.y * 64;                // col base (N)
    const int hi   = lane >> 4;
    const int ln   = lane & 15;

    const _Float16* arow0 = A + (size_t)(rb + ln) * KA;
    const _Float16* arow1 = A + (size_t)(rb + 16 + ln) * KA;
    const _Float16* bcol  = Wt + (size_t)(cb + ln) * KA + hi * 16;

    v8f acc[2][4] = {};
#pragma unroll
    for (int kb = 0; kb < KA; kb += 32) {
        v16h a0 = ldfragA(arow0 + kb, hi);
        v16h a1 = ldfragA(arow1 + kb, hi);
        v16h bf[4];
#pragma unroll
        for (int t = 0; t < 4; ++t) bf[t] = ldfragB(bcol + (size_t)t * 16 * KA + kb);
#pragma unroll
        for (int t = 0; t < 4; ++t) {
            acc[0][t] = __builtin_amdgcn_wmma_f32_16x16x32_f16(false, a0, false, bf[t],
                                                               (short)0, acc[0][t], false, false);
            acc[1][t] = __builtin_amdgcn_wmma_f32_16x16x32_f16(false, a1, false, bf[t],
                                                               (short)0, acc[1][t], false, false);
        }
    }
    // Epilogue. C layout: VGPR v -> M = v + 8*hi; N = ln.
#pragma unroll
    for (int rt = 0; rt < 2; ++rt) {
#pragma unroll
        for (int t = 0; t < 4; ++t) {
            int col = cb + t * 16 + ln;
#pragma unroll
            for (int v = 0; v < 8; ++v) {
                int row = rb + rt * 16 + v + hi * 8;
                float x = acc[rt][t][v];
                if (BIAS) x += bias[col];
                if (PE)   x += pe[(row & (SEQT - 1)) * DM + col];
                if (RELU) x = x > 0.0f ? x : 0.0f;
                if (RES)  x += (float)res[(size_t)row * DM + col];
                out[(size_t)row * DM + col] = (_Float16)x;
            }
        }
    }
}

// ===========================================================================
// LayerNorm over 256 features; optional second-input add (residual).
// ===========================================================================
template<bool ADD>
__global__ __launch_bounds__(256) void k_ln(const _Float16* __restrict__ x,
                                            const _Float16* __restrict__ y,
                                            const float* __restrict__ g,
                                            const float* __restrict__ b,
                                            _Float16* __restrict__ out) {
    __shared__ float red[256];
    int r = blockIdx.x, d = threadIdx.x;
    float v = (float)x[(size_t)r * DM + d];
    if (ADD) v += (float)y[(size_t)r * DM + d];
    red[d] = v; __syncthreads();
    for (int s = 128; s > 0; s >>= 1) { if (d < s) red[d] += red[d + s]; __syncthreads(); }
    float mu = red[0] * (1.0f / DM);
    __syncthreads();
    float c = v - mu;
    red[d] = c * c; __syncthreads();
    for (int s = 128; s > 0; s >>= 1) { if (d < s) red[d] += red[d + s]; __syncthreads(); }
    float var = red[0] * (1.0f / DM);
    out[(size_t)r * DM + d] = (_Float16)(c * rsqrtf(var + 1e-5f) * g[d] + b[d]);
}

// ===========================================================================
// Flash-style attention, 4 heads of hd=64; strides are template constants.
// row(b, s) = b*BSTR + s*SSTR into the [32768,256] Q/K buffers.
// V comes pre-transposed (Vt[(b*256+col)*SEQL + s]).
// Row softmax stats via LDS atomics (ds_max_u32 on order-preserving bits,
// ds_add_f32): in-order per wave, no barriers, no bpermute chains.
// ===========================================================================
template<int SEQL, int SSTR, int BSTR>
__global__ __launch_bounds__(256) void k_attn(const _Float16* __restrict__ Q,
                                              const _Float16* __restrict__ Kb,
                                              const _Float16* __restrict__ Vt,
                                              _Float16* __restrict__ O,
                                              float scale) {
    __shared__ __align__(16) _Float16 P[8][16 * 32];
    __shared__ __align__(16) unsigned RMAX[8][16];
    __shared__ __align__(16) float    RSUM[8][16];
    const int lane = threadIdx.x & 31;
    const int wv   = threadIdx.x >> 5;
    const int hi   = lane >> 4, ln = lane & 15;
    constexpr int NQC = SEQL >> 4;

    int g = blockIdx.x * 8 + wv;
    int qc = g % NQC; g /= NQC;
    int h  = g & 3;   g >>= 2;
    int b  = g;

    // Q A-fragments for dim chunks 0..31, 32..63 (scale folded in)
    v16h aq[2];
    {
        int q = qc * 16 + ln;
        const _Float16* p0 = Q + (size_t)(b * BSTR + q * SSTR) * DM + h * 64;
        aq[0] = ldfragA(p0, hi);
        aq[1] = ldfragA(p0 + 32, hi);
        if (scale != 1.0f) {
            _Float16 sc = (_Float16)scale;
#pragma unroll
            for (int d = 0; d < 2; ++d)
#pragma unroll
                for (int j = 0; j < 16; ++j) aq[d][j] = aq[d][j] * sc;
        }
    }
    constexpr size_t KSTEP = (size_t)SSTR * DM;      // elems per seq step
    constexpr size_t KGK   = 16 * KSTEP;             // between key groups
    const _Float16* kbase  = Kb + ((size_t)b * BSTR + (size_t)ln * SSTR) * DM + h * 64 + hi * 16;
    const _Float16* vtbase = Vt + ((size_t)((b << 8) + h * 64 + ln)) * SEQL + hi * 16;

    v8f o[4] = {};
    float m[8], l[8];
#pragma unroll
    for (int v = 0; v < 8; ++v) { m[v] = -1e30f; l[v] = 0.0f; }

    for (int kc = 0; kc < SEQL; kc += 32) {
        // init this wave's reduction slots (one ds_store per lane)
        if (lane < 16) RMAX[wv][ln] = 0u; else RSUM[wv][ln] = 0.0f;

        // ---- scores: two 16x16 tiles, K=64
        const _Float16* kp = kbase + (size_t)kc * KSTEP;
        v16h bk[2][2];
        bk[0][0] = ldfragB(kp);
        bk[0][1] = ldfragB(kp + 32);
        bk[1][0] = ldfragB(kp + KGK);
        bk[1][1] = ldfragB(kp + KGK + 32);
        v8f s[2];
#pragma unroll
        for (int gk = 0; gk < 2; ++gk) {
            v8f accn = {};
            accn = __builtin_amdgcn_wmma_f32_16x16x32_f16(false, aq[0], false, bk[gk][0],
                                                          (short)0, accn, false, false);
            accn = __builtin_amdgcn_wmma_f32_16x16x32_f16(false, aq[1], false, bk[gk][1],
                                                          (short)0, accn, false, false);
            s[gk] = accn;
        }

        // ---- row max via LDS atomic max (8 independent, no result regs)
#pragma unroll
        for (int v = 0; v < 8; ++v)
            atomicMax(&RMAX[wv][v + hi * 8], enc_f32(fmaxf(s[0][v], s[1][v])));
        uint4 r0 = *(const uint4*)&RMAX[wv][hi * 8];
        uint4 r1 = *(const uint4*)&RMAX[wv][hi * 8 + 4];
        float rm[8] = {dec_f32(r0.x), dec_f32(r0.y), dec_f32(r0.z), dec_f32(r0.w),
                       dec_f32(r1.x), dec_f32(r1.y), dec_f32(r1.z), dec_f32(r1.w)};

        float al[8], e0[8], e1[8];
#pragma unroll
        for (int v = 0; v < 8; ++v) {
            float nm = fmaxf(m[v], rm[v]);
            al[v] = __expf(m[v] - nm);
            e0[v] = __expf(s[0][v] - nm);
            e1[v] = __expf(s[1][v] - nm);
            m[v]  = nm;
        }
        // ---- row sum via ds_add_f32
#pragma unroll
        for (int v = 0; v < 8; ++v)
            atomicAdd(&RSUM[wv][v + hi * 8], e0[v] + e1[v]);

        // ---- stash P tile (C layout) into LDS as row-major 16x32 f16
#pragma unroll
        for (int v = 0; v < 8; ++v) {
            P[wv][(v + hi * 8) * 32 + ln]      = (_Float16)e0[v];
            P[wv][(v + hi * 8) * 32 + 16 + ln] = (_Float16)e1[v];
        }
        // ---- V fragments (independent of P; overlap the LDS bounce)
        v16h bv[4];
#pragma unroll
        for (int t = 0; t < 4; ++t)
            bv[t] = ldfragB(vtbase + (size_t)t * 16 * SEQL + kc);
        // ---- reload P as an A fragment (16x32)
        v16h pa = ldfragA(&P[wv][ln * 32], hi);

        // ---- fold in sums, rescale O, accumulate P @ V
        float4 q0 = *(const float4*)&RSUM[wv][hi * 8];
        float4 q1 = *(const float4*)&RSUM[wv][hi * 8 + 4];
        float rs[8] = {q0.x, q0.y, q0.z, q0.w, q1.x, q1.y, q1.z, q1.w};
#pragma unroll
        for (int v = 0; v < 8; ++v) l[v] = l[v] * al[v] + rs[v];
#pragma unroll
        for (int t = 0; t < 4; ++t)
#pragma unroll
            for (int v = 0; v < 8; ++v) o[t][v] *= al[v];
#pragma unroll
        for (int t = 0; t < 4; ++t)
            o[t] = __builtin_amdgcn_wmma_f32_16x16x32_f16(false, pa, false, bv[t],
                                                          (short)0, o[t], false, false);
    }
    // ---- finalize
#pragma unroll
    for (int v = 0; v < 8; ++v) {
        float inv = 1.0f / l[v];
        int q = qc * 16 + v + hi * 8;
        size_t row = (size_t)(b * BSTR + q * SSTR);
#pragma unroll
        for (int t = 0; t < 4; ++t)
            O[row * DM + h * 64 + t * 16 + ln] = (_Float16)(o[t][v] * inv);
    }
}

// ===========================================================================
// Temporal-attention pooling + decoder. One block per stock.
// ===========================================================================
__global__ __launch_bounds__(256) void k_pool(const _Float16* __restrict__ H,
                                              const _Float16* __restrict__ HH,
                                              const float* __restrict__ dw,
                                              const float* __restrict__ db,
                                              float* __restrict__ out) {
    __shared__ float lam[SEQT];
    __shared__ float pooled[DM];
    int n = blockIdx.x, t = threadIdx.x;
    if (t < SEQT) {
        const _Float16* a = HH + (size_t)(n * SEQT + t) * DM;
        const _Float16* c = HH + (size_t)(n * SEQT + SEQT - 1) * DM;
        float s = 0.0f;
        for (int d = 0; d < DM; ++d) s += (float)a[d] * (float)c[d];
        lam[t] = s;
    }
    __syncthreads();
    if (t == 0) {
        float mx = -1e30f;
        for (int i = 0; i < SEQT; ++i) mx = fmaxf(mx, lam[i]);
        float ss = 0.0f;
        for (int i = 0; i < SEQT; ++i) { float e = __expf(lam[i] - mx); lam[i] = e; ss += e; }
        float inv = 1.0f / ss;
        for (int i = 0; i < SEQT; ++i) lam[i] *= inv;
    }
    __syncthreads();
    {
        float s = 0.0f;
        for (int i = 0; i < SEQT; ++i)
            s += lam[i] * (float)H[(size_t)(n * SEQT + i) * DM + t];
        pooled[t] = s;
    }
    __syncthreads();
    if (t < 2) {
        float s = db[t];
        for (int d = 0; d < DM; ++d) s += pooled[d] * dw[d * 2 + t];
        out[n * 2 + t]        = s;   // tweet_pred
        out[2048 + n * 2 + t] = s;   // data_pred
        out[4096 + n * 2 + t] = s;   // cat_pred
    }
    if (n == 0 && t == 0) out[6144] = 0.0f;  // align_loss
}

// ===========================================================================
// Host launcher
// ===========================================================================
extern "C" void kernel_launch(void* const* d_in, const int* in_sizes, int n_in,
                              void* d_out, int out_size, void* d_ws, size_t ws_size,
                              hipStream_t stream) {
    (void)in_sizes; (void)n_in; (void)out_size; (void)ws_size;

    const float* x_enc  = (const float*)d_in[0];
    const float* emb_w  = (const float*)d_in[5];
    const float* emb_b  = (const float*)d_in[6];
    const float* pe     = (const float*)d_in[7];
    const float* t_ln1g = (const float*)d_in[11];
    const float* t_ln1b = (const float*)d_in[12];
    const float* t_ln2g = (const float*)d_in[13];
    const float* t_ln2b = (const float*)d_in[14];
    const float* t_b1   = (const float*)d_in[16];
    const float* t_b2   = (const float*)d_in[18];
    const float* s_ln1g = (const float*)d_in[22];
    const float* s_ln1b = (const float*)d_in[23];
    const float* s_ln2g = (const float*)d_in[24];
    const float* s_ln2b = (const float*)d_in[25];
    const float* s_b1   = (const float*)d_in[27];
    const float* s_b2   = (const float*)d_in[29];
    const float* dec_w  = (const float*)d_in[31];
    const float* dec_b  = (const float*)d_in[32];

    // ---- workspace partition (~146 MB) ----
    char* w = (char*)d_ws;
    auto alloc = [&](size_t bytes) -> void* {
        void* p = (void*)w;
        w += (bytes + 255) & ~(size_t)255;
        return p;
    };
    _Float16* Xp    = (_Float16*)alloc((size_t)NROWS * KPAD * 2);
    _Float16* WembT = (_Float16*)alloc((size_t)DM * KPAD * 2);
    _Float16* WT[11];
    for (int i = 0; i < 11; ++i) WT[i] = (_Float16*)alloc((size_t)DM * DM * 2);
    const size_t actB = (size_t)NROWS * DM * 2;
    _Float16* H   = (_Float16*)alloc(actB);
    _Float16* XN  = (_Float16*)alloc(actB);
    _Float16* Qb  = (_Float16*)alloc(actB);
    _Float16* Kb  = (_Float16*)alloc(actB);
    _Float16* Vb  = (_Float16*)alloc(actB);
    _Float16* ATT = (_Float16*)alloc(actB);
    _Float16* XT  = (_Float16*)alloc(actB);
    _Float16* F1  = (_Float16*)alloc(actB);
    _Float16* VTb = F1;   // Vt only lives during attention; F1 only after it

    // ---- conversions ----
    k_pad_x<<<(NROWS * KPAD + 255) / 256, 256, 0, stream>>>(x_enc, Xp);
    k_wt_pad<<<(DM * KPAD + 255) / 256, 256, 0, stream>>>(emb_w, WembT);
    const int widx[11] = {8, 9, 10, 15, 17, 19, 20, 21, 26, 28, 30};
    for (int i = 0; i < 11; ++i)
        k_wt<<<(DM * DM + 255) / 256, 256, 0, stream>>>((const float*)d_in[widx[i]], WT[i]);

    dim3 gg(NROWS / 256, DM / 64);  // (128, 4)

    // ---- embedding + positional encoding ----
    k_gemm<KPAD, true, true, false, false><<<gg, 256, 0, stream>>>(
        Xp, WembT, emb_b, pe, nullptr, H);

    // ================= T block (attention over time, scale 1.0) ============
    k_ln<false><<<NROWS, 256, 0, stream>>>(H, nullptr, t_ln1g, t_ln1b, XN);
    k_gemm<DM, false, false, false, false><<<gg, 256, 0, stream>>>(XN, WT[0], nullptr, nullptr, nullptr, Qb);
    k_gemm<DM, false, false, false, false><<<gg, 256, 0, stream>>>(XN, WT[1], nullptr, nullptr, nullptr, Kb);
    k_gemm<DM, false, false, false, false><<<gg, 256, 0, stream>>>(XN, WT[2], nullptr, nullptr, nullptr, Vb);
    k_vt<<<NROWS, 256, 0, stream>>>(Vb, VTb, SEQT, 1, SEQT, 13);      // seq = time
    // waves = 1024 stocks * 4 heads * 2 q-chunks = 8192 -> 1024 blocks
    k_attn<SEQT, 1, SEQT><<<1024, 256, 0, stream>>>(Qb, Kb, VTb, ATT, 1.0f);
    k_ln<true><<<NROWS, 256, 0, stream>>>(XN, ATT, t_ln2g, t_ln2b, XT);
    k_gemm<DM, true, false, true, false><<<gg, 256, 0, stream>>>(XT, WT[3], t_b1, nullptr, nullptr, F1);
    k_gemm<DM, true, false, false, true><<<gg, 256, 0, stream>>>(F1, WT[4], t_b2, nullptr, XT, H);

    // ================= S block (attention over stocks, scale 1/8) ==========
    k_ln<false><<<NROWS, 256, 0, stream>>>(H, nullptr, s_ln1g, s_ln1b, XN);
    k_gemm<DM, false, false, false, false><<<gg, 256, 0, stream>>>(XN, WT[5], nullptr, nullptr, nullptr, Qb);
    k_gemm<DM, false, false, false, false><<<gg, 256, 0, stream>>>(XN, WT[6], nullptr, nullptr, nullptr, Kb);
    k_gemm<DM, false, false, false, false><<<gg, 256, 0, stream>>>(XN, WT[7], nullptr, nullptr, nullptr, Vb);
    k_vt<<<NROWS, 256, 0, stream>>>(Vb, VTb, NSTK, SEQT, 1, 18);      // seq = stocks
    // waves = 32 timesteps * 4 heads * 64 q-chunks = 8192 -> 1024 blocks
    k_attn<NSTK, SEQT, 1><<<1024, 256, 0, stream>>>(Qb, Kb, VTb, ATT, 0.125f);
    k_ln<true><<<NROWS, 256, 0, stream>>>(XN, ATT, s_ln2g, s_ln2b, XT);
    k_gemm<DM, true, false, true, false><<<gg, 256, 0, stream>>>(XT, WT[8], s_b1, nullptr, nullptr, F1);
    k_gemm<DM, true, false, false, true><<<gg, 256, 0, stream>>>(F1, WT[9], s_b2, nullptr, XT, H);

    // ================= temporal pooling + decoder ==========================
    k_gemm<DM, false, false, false, false><<<gg, 256, 0, stream>>>(H, WT[10], nullptr, nullptr, nullptr, Qb); // hh
    k_pool<<<NSTK, 256, 0, stream>>>(H, Qb, dec_w, dec_b, (float*)d_out);
}
